// Graph_Net_18382460027234
// MI455X (gfx1250) — matrix operand order repaired and checked by reading, not compiled, not run on hardware
//
#include <hip/hip_runtime.h>
#include <hip/hip_bf16.h>

typedef __bf16 bf16;
typedef __attribute__((ext_vector_type(16))) __bf16 v16bf;
typedef __attribute__((ext_vector_type(8)))  __bf16 v8bf;
typedef __attribute__((ext_vector_type(8)))  float  v8f;

#define LDA 40   // padded LDS row stride (bf16 elems): 80B -> 20-bank stride, conflict-free
#define LDB 40

__device__ __forceinline__ v16bf cat8(v8bf lo, v8bf hi) {
  return __builtin_shufflevector(lo, hi, 0, 1, 2, 3, 4, 5, 6, 7,
                                 8, 9, 10, 11, 12, 13, 14, 15);
}

// Stage A tile (64x32) and B tile (32x128, stored transposed N x K) as bf16.
// GUARD=false: k0+32 <= K, no bounds checks -> wide batched loads.
template <bool GUARD>
__device__ __forceinline__ void stage_tiles(const float* __restrict__ A,
                                            const float* __restrict__ B,
                                            bf16* shA, bf16* shB,
                                            int tid, int row0, int M, int K, int k0) {
  // ---- A: thread -> row r = tid>>2, k-segment of 8 = (tid&3)*8
  {
    int r  = tid >> 2;
    int ks = (tid & 3) << 3;
    int gr = row0 + r;
    gr = gr < M ? gr : M - 1;  // branchless clamp (extra rows computed, never stored)
    const float* pa = A + (size_t)gr * K + k0 + ks;
    float t0[8];
#pragma unroll
    for (int j = 0; j < 8; ++j)
      t0[j] = (!GUARD || (k0 + ks + j < K)) ? pa[j] : 0.0f;
    v8bf pk;
#pragma unroll
    for (int j = 0; j < 8; ++j) pk[j] = (bf16)t0[j];
    *(v8bf*)&shA[r * LDA + ks] = pk;   // one ds_store_b128
  }
  // ---- B: thread -> col n = tid&127, k-half of 16 = (tid>>7)*16 (coalesced across lanes)
  {
    int n  = tid & 127;
    int kh = (tid >> 7) << 4;
    const float* pb = B + (size_t)(k0 + kh) * 128 + n;
    float t1[16];
#pragma unroll
    for (int j = 0; j < 16; ++j)
      t1[j] = (!GUARD || (k0 + kh + j < K)) ? pb[(size_t)j * 128] : 0.0f;
    v8bf q0, q1;
#pragma unroll
    for (int j = 0; j < 8; ++j) { q0[j] = (bf16)t1[j]; q1[j] = (bf16)t1[8 + j]; }
    *(v8bf*)&shB[n * LDB + kh]     = q0;  // two ds_store_b128
    *(v8bf*)&shB[n * LDB + kh + 8] = q1;
  }
}

// ---------------------------------------------------------------------------
// WMMA GEMM: C[M x 128] = A[M x K](f32) @ B[K x 128](f32), bf16 compute.
// Block = 256 threads = 8 wave32. Block tile 64 rows x 128 cols.
// Wave w: 16-col slab w, four 16x16 row sub-tiles -> 4 WMMAs reuse one B frag.
// ---------------------------------------------------------------------------
__global__ __launch_bounds__(256)
void gemm_bf16_n128(const float* __restrict__ A, const float* __restrict__ B,
                    float* __restrict__ C, int M, int K) {
  __shared__ bf16 shA[64 * LDA];    // 5.0 KB
  __shared__ bf16 shB[128 * LDB];   // 10.0 KB (transposed: [n][k])

  const int tid  = threadIdx.x;
  const int lane = tid & 31;
  const int wave = tid >> 5;        // col slab
  const int half = lane >> 4;
  const int mr   = lane & 15;
  const int row0 = blockIdx.x * 64;

  v8f acc[4];
#pragma unroll
  for (int t = 0; t < 4; ++t) acc[t] = v8f{0.f, 0.f, 0.f, 0.f, 0.f, 0.f, 0.f, 0.f};

  int k0 = 0;
  for (; k0 + 32 <= K; k0 += 32) {
    stage_tiles<false>(A, B, shA, shB, tid, row0, M, K, k0);
    if (k0 + 32 < K)
      __builtin_prefetch(&A[(size_t)(row0 + (tid >> 2)) * K + k0 + 32], 0, 0);
    __syncthreads();
    // B fragment: lane col = wave*16+mr, contiguous K run [half*16, +16) -> 2x ds_load_b128
    const bf16* pB = &shB[(wave * 16 + mr) * LDB + (half << 4)];
    v16bf bfrag = cat8(*(const v8bf*)pB, *(const v8bf*)(pB + 8));
#pragma unroll
    for (int t = 0; t < 4; ++t) {
      // A fragment: lane row = t*16+mr, contiguous K run [half*16, +16)
      const bf16* pA = &shA[(t * 16 + mr) * LDA + (half << 4)];
      v16bf afrag = cat8(*(const v8bf*)pA, *(const v8bf*)(pA + 8));
      acc[t] = __builtin_amdgcn_wmma_f32_16x16x32_bf16(
          false, afrag, false, bfrag, (short)0, acc[t], false, false);
    }
    __syncthreads();
  }
  if (k0 < K) {  // K tail (< 32), guarded staging, runs once
    stage_tiles<true>(A, B, shA, shB, tid, row0, M, K, k0);
    __syncthreads();
    const bf16* pB = &shB[(wave * 16 + mr) * LDB + (half << 4)];
    v16bf bfrag = cat8(*(const v8bf*)pB, *(const v8bf*)(pB + 8));
#pragma unroll
    for (int t = 0; t < 4; ++t) {
      const bf16* pA = &shA[(t * 16 + mr) * LDA + (half << 4)];
      v16bf afrag = cat8(*(const v8bf*)pA, *(const v8bf*)(pA + 8));
      acc[t] = __builtin_amdgcn_wmma_f32_16x16x32_bf16(
          false, afrag, false, bfrag, (short)0, acc[t], false, false);
    }
    __syncthreads();
  }

  // C/D layout: acc[t][r] -> row = row0 + t*16 + r + 8*half, col = wave*16 + mr
#pragma unroll
  for (int t = 0; t < 4; ++t) {
#pragma unroll
    for (int r = 0; r < 8; ++r) {
      int m = row0 + t * 16 + r + 8 * half;
      if (m < M) C[(size_t)m * 128 + wave * 16 + mr] = acc[t][r];
    }
  }
}

// ---------------------------------------------------------------------------
// Graph aggregation helpers
// ---------------------------------------------------------------------------
__global__ void zero_f32(float* __restrict__ p, int n) {
  int i = blockIdx.x * blockDim.x + threadIdx.x;
  if (i < n) p[i] = 0.0f;
}

__global__ void deg_accum(const int* __restrict__ dst, float* __restrict__ deg, int E) {
  int e = blockIdx.x * blockDim.x + threadIdx.x;
  if (e < E) atomicAdd(&deg[dst[e]], 1.0f);
}

__global__ void dinv_k(const float* __restrict__ deg, float* __restrict__ dinv, int n) {
  int i = blockIdx.x * blockDim.x + threadIdx.x;
  if (i < n) dinv[i] = rsqrtf(deg[i] + 2.0f);  // SELF_LOOP_W = 2
}

// AGG = 2 * dinv^2 * H   (self-loop contribution, improved=True)
__global__ void agg_init(const float* __restrict__ H, const float* __restrict__ dinv,
                         float* __restrict__ AGG, int total) {
  int i = blockIdx.x * blockDim.x + threadIdx.x;
  if (i < total) {
    float di = dinv[i >> 7];
    AGG[i] = 2.0f * di * di * H[i];
  }
}

// AGG[dst] += dinv[src]*dinv[dst] * H[src]; 32 lanes/edge, 4 feats/lane
__global__ void scatter_agg(const int* __restrict__ src, const int* __restrict__ dst,
                            const float* __restrict__ dinv, const float* __restrict__ H,
                            float* __restrict__ AGG, int E) {
  int t = blockIdx.x * blockDim.x + threadIdx.x;
  int e = t >> 5;
  int g = t & 31;
  if (e >= E) return;
  int s = src[e], d = dst[e];
  float norm = dinv[s] * dinv[d];
  float4 v = ((const float4*)(H + (size_t)s * 128))[g];
  float* o = AGG + (size_t)d * 128 + g * 4;
  atomicAdd(o + 0, norm * v.x);
  atomicAdd(o + 1, norm * v.y);
  atomicAdd(o + 2, norm * v.z);
  atomicAdd(o + 3, norm * v.w);
}

__global__ void bias_act(const float* __restrict__ AGG, const float* __restrict__ bias,
                         float* __restrict__ out, int total, int relu) {
  int i = blockIdx.x * blockDim.x + threadIdx.x;
  if (i < total) {
    float v = AGG[i] + bias[i & 127];
    out[i] = relu ? fmaxf(v, 0.0f) : v;
  }
}

// out = relu(XWc1 + bc1) @ Wc2 + bc2 ; also emit node_mask (eval mode -> keep)
__global__ void classifier(const float* __restrict__ XWc1, const float* __restrict__ bc1,
                           const float* __restrict__ Wc2, const float* __restrict__ bc2,
                           float* __restrict__ out, float* __restrict__ mask, int N) {
  int n = blockIdx.x * blockDim.x + threadIdx.x;
  if (n >= N) return;
  float a0 = bc2[0], a1 = bc2[1], a2 = bc2[2], a3 = bc2[3];
  const float* row = XWc1 + (size_t)n * 128;
#pragma unroll 4
  for (int k = 0; k < 128; ++k) {
    float v = fmaxf(row[k] + bc1[k], 0.0f);
    const float* w = Wc2 + k * 4;
    a0 += v * w[0]; a1 += v * w[1]; a2 += v * w[2]; a3 += v * w[3];
  }
  float* o = out + (size_t)n * 4;
  o[0] = a0; o[1] = a1; o[2] = a2; o[3] = a3;
  mask[n] = 1.0f;
}

// ---------------------------------------------------------------------------
extern "C" void kernel_launch(void* const* d_in, const int* in_sizes, int n_in,
                              void* d_out, int out_size, void* d_ws, size_t ws_size,
                              hipStream_t stream) {
  (void)n_in; (void)out_size; (void)ws_size;
  const float* x   = (const float*)d_in[0];
  const int*   ei  = (const int*)d_in[1];
  const float* W1  = (const float*)d_in[2];
  const float* b1  = (const float*)d_in[3];
  const float* W2  = (const float*)d_in[4];
  const float* b2  = (const float*)d_in[5];
  const float* Wc1 = (const float*)d_in[6];
  const float* bc1 = (const float*)d_in[7];
  const float* Wc2 = (const float*)d_in[8];
  const float* bc2 = (const float*)d_in[9];

  const int E  = in_sizes[1] / 2;        // 500000
  const int K1 = in_sizes[2] / 128;      // 5000 (IN_DIM)
  const int N  = in_sizes[0] / K1;       // 50000 nodes

  const int* src = ei;
  const int* dst = ei + E;

  // workspace layout (floats)
  float* XW   = (float*)d_ws;                  // N*128  (gemm output, reused)
  float* AGG  = XW   + (size_t)N * 128;        // N*128
  float* DINV = AGG  + (size_t)N * 128;        // N
  float* DEG  = DINV + N;                      // N
  float* H1   = DEG  + N;                      // N*128

  float* out_logits = (float*)d_out;               // N*4
  float* out_mask   = out_logits + (size_t)N * 4;  // N
  float* out_h      = out_mask + N;                // N*128

  const int feat_total = N * 128;
  dim3 blk(256);
  dim3 gN((N + 255) / 256), gE((E + 255) / 256);
  dim3 gF((feat_total + 255) / 256), gS(((size_t)E * 32 + 255) / 256);
  dim3 gG((N + 63) / 64);  // 64-row block tiles

  // degrees + normalization
  zero_f32<<<gN, blk, 0, stream>>>(DEG, N);
  deg_accum<<<gE, blk, 0, stream>>>(dst, DEG, E);
  dinv_k<<<gN, blk, 0, stream>>>(DEG, DINV, N);

  // layer 1: XW = x @ W1 ; agg ; relu(+b1)
  gemm_bf16_n128<<<gG, blk, 0, stream>>>(x, W1, XW, N, K1);
  agg_init<<<gF, blk, 0, stream>>>(XW, DINV, AGG, feat_total);
  scatter_agg<<<gS, blk, 0, stream>>>(src, dst, DINV, XW, AGG, E);
  bias_act<<<gF, blk, 0, stream>>>(AGG, b1, H1, feat_total, 1);

  // layer 2: XW = H1 @ W2 ; agg ; +b2 -> h (third output)
  gemm_bf16_n128<<<gG, blk, 0, stream>>>(H1, W2, XW, N, 128);
  agg_init<<<gF, blk, 0, stream>>>(XW, DINV, AGG, feat_total);
  scatter_agg<<<gS, blk, 0, stream>>>(src, dst, DINV, XW, AGG, E);
  bias_act<<<gF, blk, 0, stream>>>(AGG, b2, out_h, feat_total, 0);

  // classifier: XW = h @ Wc1 ; out = relu(XW+bc1) @ Wc2 + bc2
  gemm_bf16_n128<<<gG, blk, 0, stream>>>(out_h, Wc1, XW, N, 128);
  classifier<<<gN, blk, 0, stream>>>(XW, bc1, Wc2, bc2, out_logits, out_mask, N);
}
